// AGFLayer_26362509263642
// MI455X (gfx1250) — matrix operand order repaired and checked by reading, not compiled, not run on hardware
//
#include <hip/hip_runtime.h>
#include <hip/hip_bf16.h>

#define B_   4
#define N_   2048
#define D_   256
#define H_   8
#define HD_  32
#define KORD 3
#define BN_  (B_ * N_)

#define TLD  2056          // LDS pitch (elements) for transposed t in spmm
#define WLD  264           // LDS pitch (elements) for W^T slice in gemm
#define SLD  2052          // LDS pitch (floats) for score rows

typedef __attribute__((ext_vector_type(16))) __bf16 v16bf;
typedef __attribute__((ext_vector_type(8)))  __bf16 v8bf;
typedef __attribute__((ext_vector_type(8)))  float  v8f;

static __device__ __forceinline__ int lane_id() { return (int)(threadIdx.x & 31); }

// ---------------------------------------------------------------------------
// WMMA fragment helpers (CDNA5 wave32 layouts, ISA 7.12.2). All loads are
// 16B v8bf so only 16B alignment is required (pitches are 16B multiples).
// ---------------------------------------------------------------------------

// A-matrix fragment: 16x32 bf16, row-major, stride lda (elements).
// lane<16 -> row=lane, K={0..7,16..23}; lane>=16 -> row=lane-16, K={8..15,24..31}
static __device__ __forceinline__ v16bf load_frag_a(const __bf16* __restrict__ base, int lda) {
  const int l  = lane_id();
  const int m  = l & 15;
  const int kb = (l >> 4) << 3;                 // 0 or 8
  const __bf16* p = base + (size_t)m * lda + kb;
  v8bf lo = *(const v8bf*)(p);                  // K = kb .. kb+7
  v8bf hi = *(const v8bf*)(p + 16);             // K = kb+16 .. kb+23
  v16bf a;
#pragma unroll
  for (int i = 0; i < 8; ++i) { a[i] = lo[i]; a[8 + i] = hi[i]; }
  return a;
}

// B-matrix fragment: 32x16, from TRANSPOSED storage BT[n][k], pitch ldt.
// lane<16 -> col=lane, K={0..7,16..23} ; lane>=16 -> same col, K={8..15,24..31}.
static __device__ __forceinline__ v16bf load_frag_bT(const __bf16* __restrict__ baseT, int ldt) {
  const int l  = lane_id();
  const int n  = l & 15;
  const int kb = (l >> 4) << 3;                 // 0 or 8
  const __bf16* p = baseT + (size_t)n * ldt + kb;
  v8bf lo = *(const v8bf*)(p);
  v8bf hi = *(const v8bf*)(p + 16);
  v16bf b;
#pragma unroll
  for (int i = 0; i < 8; ++i) { b[i] = lo[i]; b[8 + i] = hi[i]; }
  return b;
}

static __device__ __forceinline__ v8f wmma_bf16(v16bf a, v16bf b, v8f c) {
  return __builtin_amdgcn_wmma_f32_16x16x32_bf16(false, a, false, b, (short)0, c, false, false);
}

// ---------------------------------------------------------------------------
// f32 -> bf16 conversion (plain, and transposing variant for weights)
// ---------------------------------------------------------------------------
__global__ void agf_cvt_bf16(const float* __restrict__ src, __bf16* __restrict__ dst, int n) {
  int i = blockIdx.x * blockDim.x + threadIdx.x;
  if (i < n) dst[i] = (__bf16)src[i];
}

// W[k][n] (DxD f32) -> WT[n][k] (DxD bf16)
__global__ void agf_cvt_wT_bf16(const float* __restrict__ W, __bf16* __restrict__ WT) {
  int i = blockIdx.x * blockDim.x + threadIdx.x;
  if (i < D_ * D_) {
    int k = i >> 8, n = i & (D_ - 1);
    WT[n * D_ + k] = (__bf16)W[i];
  }
}

// ---------------------------------------------------------------------------
// GEMM vs pre-transposed weight: C[M,256] = A[M,256] * W (+bias).
// WG = 256 threads / 8 waves; block covers 128 rows x 32 cols.
// W^T slice (32 cols x 256 k, 16KB) staged ONCE in LDS -> barrier-free k loop:
// per iter 1 global A-frag -> 2 WMMA.
// ---------------------------------------------------------------------------
__global__ void agf_gemm_xw(const __bf16* __restrict__ A, const __bf16* __restrict__ WT,
                            const float* __restrict__ bias, float* __restrict__ Cf,
                            __bf16* __restrict__ Cb) {
  __shared__ __align__(16) __bf16 Bt[32 * WLD];  // WT slice [n=32][k=256]
  const int tid  = threadIdx.x;                  // 256
  const int wave = tid >> 5;
  const int row0 = blockIdx.x * 128;
  const int col0 = blockIdx.y * 32;

  {
    const int n = tid >> 3, k0 = (tid & 7) * 32;
#pragma unroll
    for (int i = 0; i < 4; ++i)
      *(v8bf*)(&Bt[n * WLD + k0 + i * 8]) =
          *(const v8bf*)(WT + (size_t)(col0 + n) * D_ + k0 + i * 8);
  }
  __syncthreads();

  const int wr = row0 + wave * 16;               // 16 rows per wave
  v8f acc0 = {}, acc1 = {};
#pragma unroll
  for (int k0 = 0; k0 < D_; k0 += 32) {
    v16bf af = load_frag_a(A + (size_t)wr * D_ + k0, D_);
    v16bf b0 = load_frag_bT(&Bt[k0], WLD);
    v16bf b1 = load_frag_bT(&Bt[16 * WLD + k0], WLD);
    acc0 = wmma_bf16(af, b0, acc0);
    acc1 = wmma_bf16(af, b1, acc1);
  }

  const int l   = lane_id();
  const int cn  = l & 15;
  const int rr0 = wr + ((l >> 4) << 3);
#pragma unroll
  for (int t = 0; t < 2; ++t) {
    v8f acc = t ? acc1 : acc0;
    const int cc = col0 + t * 16 + cn;
    const float bcv = bias ? bias[cc] : 0.0f;
#pragma unroll
    for (int r = 0; r < 8; ++r) {
      float v = acc[r] + bcv;
      size_t idx = (size_t)(rr0 + r) * D_ + cc;
      if (Cf) Cf[idx] = v;
      if (Cb) Cb[idx] = (__bf16)v;
    }
  }
}

// ---------------------------------------------------------------------------
// Fused scores + softmax: A16[b,h,n,:] = softmax(q.kT/sqrt(hd)) in bf16.
// WG = 16 query rows; 16xN_ scores in dynamic LDS (pitch SLD). K rows are
// already the transposed-B layout (hd contiguous) -> barrier-free WMMA loop,
// 2 WMMA per iteration with the Q fragment held in registers.
// ---------------------------------------------------------------------------
__global__ void agf_scores_softmax(const __bf16* __restrict__ Q,
                                   const __bf16* __restrict__ Kt,
                                   __bf16* __restrict__ A16) {
  extern __shared__ float S[];                   // [16][SLD]
  __shared__ float redm[16][16];
  __shared__ float reds[16][16];
  const int n0   = blockIdx.x * 16;
  const int h    = blockIdx.y;
  const int b    = blockIdx.z;
  const int tid  = threadIdx.x;                  // 256 = 8 waves
  const int wave = tid >> 5;
  const float scale = 0.17677669529663687f;      // 1/sqrt(32)

  v16bf qf = load_frag_a(Q + ((size_t)(b * N_ + n0)) * D_ + h * HD_, D_);
  const int l  = lane_id();
  const int cn = l & 15;
  const int rb = (l >> 4) << 3;
#pragma unroll 2
  for (int j = 0; j < 8; ++j) {
    const int key0 = wave * 256 + j * 32;
    v16bf kf0 = load_frag_bT(Kt + ((size_t)(b * N_ + key0)) * D_ + h * HD_, D_);
    v16bf kf1 = load_frag_bT(Kt + ((size_t)(b * N_ + key0 + 16)) * D_ + h * HD_, D_);
    v8f s0 = {}, s1 = {};
    s0 = wmma_bf16(qf, kf0, s0);
    s1 = wmma_bf16(qf, kf1, s1);
#pragma unroll
    for (int r = 0; r < 8; ++r) {
      S[(rb + r) * SLD + key0 + cn]      = s0[r] * scale;
      S[(rb + r) * SLD + key0 + 16 + cn] = s1[r] * scale;
    }
  }
  __syncthreads();

  const int row = tid >> 4, sub = tid & 15;      // 16 threads per row
  float m = -3.0e38f;
  for (int i = sub; i < N_; i += 16) m = fmaxf(m, S[row * SLD + i]);
  redm[row][sub] = m;
  __syncthreads();
  m = redm[row][0];
#pragma unroll
  for (int i = 1; i < 16; ++i) m = fmaxf(m, redm[row][i]);

  float psum = 0.0f;
  for (int i = sub; i < N_; i += 16) {
    float e = __expf(S[row * SLD + i] - m);
    S[row * SLD + i] = e;
    psum += e;
  }
  reds[row][sub] = psum;
  __syncthreads();
  float ssum = 0.0f;
#pragma unroll
  for (int i = 0; i < 16; ++i) ssum += reds[row][i];
  const float inv = 1.0f / ssum;

  __bf16* arow = A16 + (((size_t)(b * H_ + h)) * N_ + (n0 + row)) * N_;
  for (int i = sub; i < N_; i += 16)
    arow[i] = (__bf16)(S[row * SLD + i] * inv);
}

// ---------------------------------------------------------------------------
// t_next = A * t ; out32 (+)= c[h,kk] * t_next.
// Whole transposed t for this (b,h) (32 x 2048 bf16, ~128.5KB LDS) staged
// once -> barrier-free K loop. Each wave: 32 rows, 2 A-frags x 2 B-frags
// = 4 WMMA per iteration (B-frags reused across row frags).
// kk==1 folds in the c0*v init; last pass also emits bf16 merged.
// ---------------------------------------------------------------------------
__global__ void agf_spmm_filter(const __bf16* __restrict__ A16, const __bf16* __restrict__ Tin,
                                __bf16* __restrict__ Tout, float* __restrict__ out32,
                                __bf16* __restrict__ mergedbf, const float* __restrict__ coeffs,
                                int kk) {
  extern __shared__ __bf16 tT[];                 // [HD_][TLD]
  const int n0   = blockIdx.x * 256;
  const int h    = blockIdx.y;
  const int b    = blockIdx.z;
  const int tid  = threadIdx.x;                  // 256 = 8 waves
  const int wave = tid >> 5;

  // stage transposed t: tT[d][m] for all m in [0,2048)
  for (int c = tid; c < (N_ * HD_) / 8; c += 256) {
    const int m  = c >> 2;
    const int d0 = (c & 3) * 8;
    v8bf tv = *(const v8bf*)(Tin + ((size_t)(b * N_ + m)) * D_ + h * HD_ + d0);
#pragma unroll
    for (int i = 0; i < 8; ++i) tT[(d0 + i) * TLD + m] = tv[i];
  }
  __syncthreads();

  const __bf16* Abh = A16 + ((size_t)(b * H_ + h)) * N_ * N_;
  const int wr = n0 + wave * 32;                 // 32 rows per wave
  v8f a00 = {}, a01 = {}, a10 = {}, a11 = {};
#pragma unroll 2
  for (int m0 = 0; m0 < N_; m0 += 32) {
    v16bf af0 = load_frag_a(Abh + (size_t)wr * N_ + m0, N_);
    v16bf af1 = load_frag_a(Abh + (size_t)(wr + 16) * N_ + m0, N_);
    v16bf b0  = load_frag_bT(&tT[m0], TLD);             // d = 0..15
    v16bf b1  = load_frag_bT(&tT[16 * TLD + m0], TLD);  // d = 16..31
    a00 = wmma_bf16(af0, b0, a00);
    a01 = wmma_bf16(af0, b1, a01);
    a10 = wmma_bf16(af1, b0, a10);
    a11 = wmma_bf16(af1, b1, a11);
  }

  const float ck = coeffs[h * (KORD + 1) + kk];
  const float c0 = coeffs[h * (KORD + 1)];
  const int l  = lane_id();
  const int cn = l & 15;
#pragma unroll
  for (int rt = 0; rt < 2; ++rt) {
    const int r0 = wr + rt * 16 + ((l >> 4) << 3);
#pragma unroll
    for (int t = 0; t < 2; ++t) {
      v8f acc = rt ? (t ? a11 : a10) : (t ? a01 : a00);
      const int d = t * 16 + cn;
#pragma unroll
      for (int r = 0; r < 8; ++r) {
        size_t idx = ((size_t)(b * N_ + r0 + r)) * D_ + h * HD_ + d;
        Tout[idx] = (__bf16)acc[r];
        float base = (kk == 1) ? c0 * (float)Tin[idx] : out32[idx];
        float o = base + ck * acc[r];
        out32[idx] = o;
        if (mergedbf) mergedbf[idx] = (__bf16)o;
      }
    }
  }
}

// ---------------------------------------------------------------------------
// Host launcher
// ---------------------------------------------------------------------------
extern "C" void kernel_launch(void* const* d_in, const int* in_sizes, int n_in,
                              void* d_out, int out_size, void* d_ws, size_t ws_size,
                              hipStream_t stream) {
  (void)in_sizes; (void)n_in; (void)out_size; (void)ws_size;
  const float* x      = (const float*)d_in[0];
  const float* Wq     = (const float*)d_in[1];
  const float* bq     = (const float*)d_in[2];
  const float* Wk     = (const float*)d_in[3];
  const float* bk     = (const float*)d_in[4];
  const float* Wv     = (const float*)d_in[5];
  const float* bv     = (const float*)d_in[6];
  const float* Wo     = (const float*)d_in[7];
  const float* bo     = (const float*)d_in[8];
  const float* coeffs = (const float*)d_in[9];
  float* out = (float*)d_out;

  char* w = (char*)d_ws;
  size_t off = 0;
  auto take = [&](size_t bytes) -> char* {
    char* p = w + off;
    off += (bytes + 255) & ~(size_t)255;
    return p;
  };
  __bf16* xbf   = (__bf16*)take((size_t)BN_ * D_ * 2);
  __bf16* wqT   = (__bf16*)take((size_t)D_ * D_ * 2);
  __bf16* wkT   = (__bf16*)take((size_t)D_ * D_ * 2);
  __bf16* wvT   = (__bf16*)take((size_t)D_ * D_ * 2);
  __bf16* woT   = (__bf16*)take((size_t)D_ * D_ * 2);
  __bf16* qbf   = (__bf16*)take((size_t)BN_ * D_ * 2);
  __bf16* kbf   = (__bf16*)take((size_t)BN_ * D_ * 2);
  __bf16* vbf   = (__bf16*)take((size_t)BN_ * D_ * 2);           // t0 / t2
  __bf16* a16   = (__bf16*)take((size_t)B_ * H_ * N_ * N_ * 2);  // 256 MB
  __bf16* tb    = (__bf16*)take((size_t)BN_ * D_ * 2);           // t1 / t3
  float*  out32 = (float*)take((size_t)BN_ * D_ * 4);
  __bf16* mrg   = (__bf16*)take((size_t)BN_ * D_ * 2);

  // 1) f32 -> bf16 conversions (weights pre-transposed)
  agf_cvt_bf16<<<(BN_ * D_ + 255) / 256, 256, 0, stream>>>(x, xbf, BN_ * D_);
  agf_cvt_wT_bf16<<<(D_ * D_ + 255) / 256, 256, 0, stream>>>(Wq, wqT);
  agf_cvt_wT_bf16<<<(D_ * D_ + 255) / 256, 256, 0, stream>>>(Wk, wkT);
  agf_cvt_wT_bf16<<<(D_ * D_ + 255) / 256, 256, 0, stream>>>(Wv, wvT);
  agf_cvt_wT_bf16<<<(D_ * D_ + 255) / 256, 256, 0, stream>>>(Wo, woT);

  // 2) projections (WMMA GEMM vs L DS-resident W^T, bf16 out)
  dim3 gproj(BN_ / 128, D_ / 32);
  agf_gemm_xw<<<gproj, 256, 0, stream>>>(xbf, wqT, bq, nullptr, qbf);
  agf_gemm_xw<<<gproj, 256, 0, stream>>>(xbf, wkT, bk, nullptr, kbf);
  agf_gemm_xw<<<gproj, 256, 0, stream>>>(xbf, wvT, bv, nullptr, vbf);

  // 3) fused scores + softmax -> A (bf16)
  dim3 gattn(N_ / 16, H_, B_);
  agf_scores_softmax<<<gattn, 256, 16 * SLD * (int)sizeof(float), stream>>>(qbf, kbf, a16);

  // 4) polynomial filter: out = sum_k c_k A^k v  (c0 folded into pass 1)
  dim3 gsp(N_ / 256, H_, B_);
  const int spds = HD_ * TLD * 2;
  agf_spmm_filter<<<gsp, 256, spds, stream>>>(a16, vbf, tb, out32, nullptr, coeffs, 1);
  agf_spmm_filter<<<gsp, 256, spds, stream>>>(a16, tb, vbf, out32, nullptr, coeffs, 2);
  agf_spmm_filter<<<gsp, 256, spds, stream>>>(a16, vbf, tb, out32, mrg, coeffs, 3);

  // 5) output projection: out = merged @ Wo + bo (f32 out)
  agf_gemm_xw<<<gproj, 256, 0, stream>>>(mrg, woT, bo, out, nullptr);
}
